// RepformerLayer_79955111182411
// MI455X (gfx1250) — matrix operand order repaired and checked by reading, not compiled, not run on hardware
//
#include <hip/hip_runtime.h>
#include <hip/hip_bf16.h>
#include <math.h>

#define NF_    2
#define NLOC_  4096
#define NNEI_  64
#define NALL_  6144
#define NG1_   128
#define NG2_   32
#define ND_    32
#define NH_    4
#define ATTNW_SHIFT_ 20.0f

typedef __attribute__((ext_vector_type(16))) _Float16 v16h;
typedef __attribute__((ext_vector_type(8)))  _Float16 v8h;
typedef __attribute__((ext_vector_type(8)))  float    v8f;

// ---------------------------------------------------------------------------
// WMMA helpers (gfx1250, wave32).  D(16x16,f32) = A(16x32,f16) * B(32x16,f16) + C
// ---------------------------------------------------------------------------
__device__ inline v8f wmma_f16(v16h a, v16h b, v8f c) {
  return __builtin_amdgcn_wmma_f32_16x16x32_f16(
      /*neg_a=*/false, a, /*neg_b=*/false, b,
      /*c_mod=*/(short)0, c, /*reuse_a=*/false, /*reuse_b=*/false);
}

__device__ inline v8f zero8() {
  v8f z = {0.f, 0.f, 0.f, 0.f, 0.f, 0.f, 0.f, 0.f};
  return z;
}

// A fragment from row-major [M][K] f16 source (tile at row0, k0).
// ISA layout (16-bit A 16x32): lanes 0-15 -> M=lane, K chunks [k0+0..7] and [k0+16..23];
// lanes 16-31 -> M=lane-16, K chunks [k0+8..15] and [k0+24..31].
// Both 8-half chunks are 16B-aligned -> single b128 loads.
__device__ inline v16h load_A_f16(const _Float16* __restrict__ base, int ld,
                                  int row0, int k0, int lane) {
  const int m  = row0 + (lane & 15);
  const int hi = (lane >> 4) & 1;
  const _Float16* p = base + (size_t)m * ld + k0 + hi * 8;
  const v8h a0 = *(const v8h*)(p);
  const v8h a1 = *(const v8h*)(p + 16);
  return __builtin_shufflevector(a0, a1, 0, 1, 2, 3, 4, 5, 6, 7,
                                         8, 9, 10, 11, 12, 13, 14, 15);
}

// B fragment where B = X^T, X stored row-major [N][K] f16 (tile at k0, n0).
// ISA layout (16-bit B 32x16): lanes 0-15 -> N=lane, K chunks [k0+0..7] and [k0+16..23];
// lanes 16-31 -> N=lane-16, K chunks [k0+8..15] and [k0+24..31].
__device__ inline v16h load_Bt_f16(const _Float16* __restrict__ base, int ld,
                                   int k0, int n0, int lane) {
  const int n  = n0 + (lane & 15);
  const int hi = (lane >> 4) & 1;
  const _Float16* p = base + (size_t)n * ld + k0 + hi * 8;
  const v8h b0 = *(const v8h*)(p);
  const v8h b1 = *(const v8h*)(p + 16);
  return __builtin_shufflevector(b0, b1, 0, 1, 2, 3, 4, 5, 6, 7,
                                         8, 9, 10, 11, 12, 13, 14, 15);
}

// ---------------------------------------------------------------------------
// Kernel 0: transpose + f16-convert the shared weight matrices into d_ws.
//   ws[0      .. 8191 ]  : wqkT[j][d] = w_qk[d][j]   (256 x 32)
//   ws[8192   .. 12287]  : wvT [j][d] = w_v [d][j]   (128 x 32)
//   ws[12288  .. 16383]  : whmT[o][c] = w_hm[c][o]   (32  x 128)
// ---------------------------------------------------------------------------
__global__ void convert_weights_kernel(const float* __restrict__ w_qk,
                                       const float* __restrict__ w_v,
                                       const float* __restrict__ w_hm,
                                       _Float16* __restrict__ ws) {
  const int t = blockIdx.x * blockDim.x + threadIdx.x;
  if (t < 8192) {
    const int j = t >> 5, d = t & 31;
    ws[t] = (_Float16)w_qk[d * 256 + j];
  } else if (t < 12288) {
    const int u = t - 8192;
    const int j = u >> 5, d = u & 31;
    ws[t] = (_Float16)w_v[d * 128 + j];
  } else if (t < 16384) {
    const int u = t - 12288;
    const int o = u >> 7, c = u & 127;
    ws[t] = (_Float16)w_hm[c * 32 + o];
  }
}

// ---------------------------------------------------------------------------
// Kernel 1: gg1 gather (pure bandwidth).  One 32-lane group per (b,l,n) row,
// one float4 per lane (128 floats/row), fully coalesced.
// ---------------------------------------------------------------------------
__global__ __launch_bounds__(256) void gather_gg1_kernel(
    const float* __restrict__ g1_ext, const float* __restrict__ sw,
    const int* __restrict__ nlist, const unsigned char* __restrict__ mask,
    float* __restrict__ out) {
  const int r    = blockIdx.x * 8 + (threadIdx.x >> 5);
  const int lane = threadIdx.x & 31;
  const int b    = r >> 18;                       // r / (NLOC_*NNEI_) = r / 262144
  const int idx  = nlist[r];
  const float s  = mask[r] ? sw[r] : 0.f;
  const float4* src = (const float4*)(g1_ext + ((size_t)b * NALL_ + idx) * NG1_);
  const float4 v = src[lane];
  float4 o;
  o.x = v.x * s; o.y = v.y * s; o.z = v.z * s; o.w = v.w * s;
  ((float4*)(out + (size_t)r * NG1_))[lane] = o;
}

// ---------------------------------------------------------------------------
// Kernel 2: attention block.  One workgroup = one (b,l); 128 threads = 4 waves.
// Wave w owns M-tile (neighbor rows 16w..16w+15) in every GEMM; all 4 waves
// cooperate on one head at a time.  ~58.5 KB static LDS.
// ---------------------------------------------------------------------------
struct SmemT {
  alignas(16) _Float16 g2[64 * 32];     // g2 tile, f16, row-major [n][kk]
  alignas(16) _Float16 q [64 * 32];     // q_h  [n][d]
  alignas(16) _Float16 k [64 * 32];     // k_h  [n][d]   (B loaded transposed)
  alignas(16) _Float16 vT[32 * 64];     // v_h^T [c][m]
  alignas(16) _Float16 AA[64 * 64];     // AA_h [n][m]
  alignas(16) _Float16 mh[64 * 128];    // mh   [n][c*4+h]
  alignas(16) float    attn[64 * 64];   // attention scores f32
  alignas(16) float    h2[64 * 4];      // h2 padded [n][x]
  alignas(16) float    sw[64];
  alignas(16) float    maskf[64];
  alignas(16) float    ev[64 * 4];      // accumulated h2_out
};

__global__ __launch_bounds__(128) void repformer_attn_kernel(
    const float* __restrict__ g2, const float* __restrict__ h2,
    const float* __restrict__ sw, const unsigned char* __restrict__ mask,
    const float* __restrict__ b_hm, const float* __restrict__ w_ev,
    const _Float16* __restrict__ wqkT, const _Float16* __restrict__ wvT,
    const _Float16* __restrict__ whmT,
    float* __restrict__ out_g2, float* __restrict__ out_h2) {
  __shared__ SmemT sm;

  const int tid  = threadIdx.x;
  const int lane = tid & 31;
  const int w    = tid >> 5;          // wave id = M-tile id
  const int wg   = blockIdx.x;        // flat (b*NLOC + l)
  const int m0   = 16 * w;
  const int hi8  = (lane & 16) ? 8 : 0;
  const int lc   = lane & 15;

  // ---- Phase A: stage g2 (f32->f16), h2, sw, mask into LDS --------------
  {
    const float* g2p = g2 + (size_t)wg * NNEI_ * NG2_;
    for (int i = tid; i < NNEI_ * NG2_; i += 128) sm.g2[i] = (_Float16)g2p[i];
    const float* h2p = h2 + (size_t)wg * NNEI_ * 3;
    for (int i = tid; i < NNEI_ * 3; i += 128) sm.h2[(i / 3) * 4 + (i % 3)] = h2p[i];
    for (int i = tid; i < NNEI_; i += 128) {
      sm.sw[i]    = sw[(size_t)wg * NNEI_ + i];
      sm.maskf[i] = mask[(size_t)wg * NNEI_ + i] ? 1.f : 0.f;
    }
    for (int i = tid; i < NNEI_ * 4; i += 128) sm.ev[i] = 0.f;
  }
  __syncthreads();

  const float rsqd = 0.17677669529663689f;   // 1/sqrt(ND)
  const float rs3  = 0.57735026918962576f;   // 1/sqrt(3)

  for (int h = 0; h < NH_; ++h) {
    // ---- C1: q_h, k_h, v_h = g2 @ W  (A = g2 tile, B from global f16 wts)
    {
      const v16h a = load_A_f16(sm.g2, NG2_, m0, 0, lane);
#pragma unroll
      for (int jt = 0; jt < 2; ++jt) {
        const int n0 = jt * 16;
        const v16h bq = load_Bt_f16(wqkT + h * 32,          8 * 32, 0, n0, lane);
        const v8f  cq = wmma_f16(a, bq, zero8());
        const v16h bk = load_Bt_f16(wqkT + (NH_ + h) * 32,  8 * 32, 0, n0, lane);
        const v8f  ck = wmma_f16(a, bk, zero8());
        const v16h bv = load_Bt_f16(wvT + h * 32,           4 * 32, 0, n0, lane);
        const v8f  cv = wmma_f16(a, bv, zero8());
#pragma unroll
        for (int r = 0; r < 8; ++r) {
          const int row = m0 + r + hi8;
          const int col = n0 + lc;
          sm.q[row * 32 + col]  = (_Float16)cq[r];
          sm.k[row * 32 + col]  = (_Float16)ck[r];
          sm.vT[col * 64 + row] = (_Float16)cv[r];
        }
      }
    }
    __syncthreads();

    // ---- C2: attn = q @ k^T / sqrt(d), fused gate / shift / sw ----------
    {
      const v16h a = load_A_f16(sm.q, 32, m0, 0, lane);
#pragma unroll
      for (int jt = 0; jt < 4; ++jt) {
        const int c0   = jt * 16;
        const v16h bK  = load_Bt_f16(sm.k, 32, 0, c0, lane);
        const v8f  c   = wmma_f16(a, bK, zero8());
        const int mcol = c0 + lc;
        const float h2m0 = sm.h2[mcol * 4 + 0];
        const float h2m1 = sm.h2[mcol * 4 + 1];
        const float h2m2 = sm.h2[mcol * 4 + 2];
        const float swm  = sm.sw[mcol];
#pragma unroll
        for (int r = 0; r < 8; ++r) {
          const int row = m0 + r + hi8;
          const float gate = sm.h2[row * 4 + 0] * h2m0 +
                             sm.h2[row * 4 + 1] * h2m1 +
                             sm.h2[row * 4 + 2] * h2m2;
          const float a2 = (c[r] * rsqd * gate + ATTNW_SHIFT_) *
                               (sm.sw[row] * swm) - ATTNW_SHIFT_;
          sm.attn[row * 64 + mcol] = a2;
        }
      }
    }
    __syncthreads();

    // ---- C3: row softmax + masks + sw2 + gate/sqrt3 -> AA ; fused ev ----
    if (tid < 64) {
      const int n = tid;
      float* rowp = &sm.attn[n * 64];
      float mx = rowp[0];
#pragma unroll 4
      for (int m = 1; m < 64; ++m) mx = fmaxf(mx, rowp[m]);
      float s = 0.f;
#pragma unroll 4
      for (int m = 0; m < 64; ++m) { const float e = expf(rowp[m] - mx); rowp[m] = e; s += e; }
      const float inv  = 1.f / s;
      const float rown = sm.maskf[n] * sm.sw[n];
      const float h2n0 = sm.h2[n * 4 + 0], h2n1 = sm.h2[n * 4 + 1], h2n2 = sm.h2[n * 4 + 2];
      float evx = 0.f, evy = 0.f, evz = 0.f;
#pragma unroll 4
      for (int m = 0; m < 64; ++m) {
        const float h2m0 = sm.h2[m * 4 + 0], h2m1 = sm.h2[m * 4 + 1], h2m2 = sm.h2[m * 4 + 2];
        const float gate = h2n0 * h2m0 + h2n1 * h2m1 + h2n2 * h2m2;
        const float aa = rowp[m] * inv * rown * sm.maskf[m] * sm.sw[m] * gate * rs3;
        sm.AA[n * 64 + m] = (_Float16)aa;
        evx += aa * h2m0; evy += aa * h2m1; evz += aa * h2m2;
      }
      const float wevh = w_ev[h];
      sm.ev[n * 4 + 0] += wevh * evx;
      sm.ev[n * 4 + 1] += wevh * evy;
      sm.ev[n * 4 + 2] += wevh * evz;
    }
    __syncthreads();

    // ---- C4: mh_h = AA_h @ v_h  (K = 64, two WMMA steps) ----------------
    {
      v8f c0 = zero8(), c1 = zero8();
#pragma unroll
      for (int kt = 0; kt < 2; ++kt) {
        const v16h a  = load_A_f16(sm.AA, 64, m0, kt * 32, lane);
        const v16h b0 = load_Bt_f16(sm.vT, 64, kt * 32, 0, lane);
        c0 = wmma_f16(a, b0, c0);
        const v16h b1 = load_Bt_f16(sm.vT, 64, kt * 32, 16, lane);
        c1 = wmma_f16(a, b1, c1);
      }
#pragma unroll
      for (int r = 0; r < 8; ++r) {
        const int row = m0 + r + hi8;
        sm.mh[row * 128 + (lc)      * 4 + h] = (_Float16)c0[r];
        sm.mh[row * 128 + (16 + lc) * 4 + h] = (_Float16)c1[r];
      }
    }
    __syncthreads();
  }

  // ---- Phase D: g2_out = mh @ w_hm + b_hm  (K = 128, four WMMA steps) ----
  {
    v8f c0 = zero8(), c1 = zero8();
#pragma unroll
    for (int kt = 0; kt < 4; ++kt) {
      const v16h a  = load_A_f16(sm.mh, 128, m0, kt * 32, lane);
      const v16h b0 = load_Bt_f16(whmT, 128, kt * 32, 0, lane);
      c0 = wmma_f16(a, b0, c0);
      const v16h b1 = load_Bt_f16(whmT, 128, kt * 32, 16, lane);
      c1 = wmma_f16(a, b1, c1);
    }
    const size_t baseo = (size_t)wg * NNEI_ * NG2_;
#pragma unroll
    for (int r = 0; r < 8; ++r) {
      const int row = m0 + r + hi8;
      out_g2[baseo + row * 32 + lc]      = c0[r] + b_hm[lc];
      out_g2[baseo + row * 32 + 16 + lc] = c1[r] + b_hm[16 + lc];
    }
  }

  // ---- Phase E: h2_out -----------------------------------------------------
  if (tid < 64) {
    const size_t baseh = ((size_t)wg * NNEI_ + tid) * 3;
    out_h2[baseh + 0] = sm.ev[tid * 4 + 0];
    out_h2[baseh + 1] = sm.ev[tid * 4 + 1];
    out_h2[baseh + 2] = sm.ev[tid * 4 + 2];
  }
}

// ---------------------------------------------------------------------------
extern "C" void kernel_launch(void* const* d_in, const int* in_sizes, int n_in,
                              void* d_out, int out_size, void* d_ws, size_t ws_size,
                              hipStream_t stream) {
  (void)in_sizes; (void)n_in; (void)out_size; (void)ws_size;
  const float* g1_ext = (const float*)d_in[0];
  const float* g2     = (const float*)d_in[1];
  const float* h2     = (const float*)d_in[2];
  const float* sw     = (const float*)d_in[3];
  const float* w_qk   = (const float*)d_in[4];
  const float* w_v    = (const float*)d_in[5];
  const float* w_hm   = (const float*)d_in[6];
  const float* b_hm   = (const float*)d_in[7];
  const float* w_ev   = (const float*)d_in[8];
  const int*   nlist  = (const int*)d_in[9];
  const unsigned char* mask = (const unsigned char*)d_in[10];

  float* out      = (float*)d_out;
  float* out_gg1  = out;
  float* out_g2   = out + (size_t)NF_ * NLOC_ * NNEI_ * NG1_;
  float* out_h2   = out_g2 + (size_t)NF_ * NLOC_ * NNEI_ * NG2_;

  _Float16* wks = (_Float16*)d_ws;   // 16384 halves = 32 KB

  convert_weights_kernel<<<64, 256, 0, stream>>>(w_qk, w_v, w_hm, wks);

  const int nrows = NF_ * NLOC_ * NNEI_;           // 524288
  gather_gg1_kernel<<<nrows / 8, 256, 0, stream>>>(g1_ext, sw, nlist, mask, out_gg1);

  repformer_attn_kernel<<<NF_ * NLOC_, 128, 0, stream>>>(
      g2, h2, sw, mask, b_hm, w_ev,
      wks, wks + 8192, wks + 12288,
      out_g2, out_h2);
}
